// Frequencyattention_16870631539407
// MI455X (gfx1250) — matrix operand order repaired and checked by reading, not compiled, not run on hardware
//
#include <hip/hip_runtime.h>
#include <hip/hip_bf16.h>

// CDNA5 / gfx1250: wave32, WMMA 16x16x32 bf16 -> f32
typedef __attribute__((ext_vector_type(16))) __bf16 v16bf;
typedef __attribute__((ext_vector_type(8)))  float  v8f;
typedef __attribute__((__vector_size__(16))) int    v4i32;   // matches builtin param type

#define S_DIM 1024
#define NT 23
#define B_DIM 4
#define HD 64

// ---- CDNA5 async global->LDS copy (ASYNCcnt), feature-gated ---------------
#if defined(__has_builtin)
# if __has_builtin(__builtin_amdgcn_global_load_async_to_lds_b128)
#  define HAVE_ASYNC_LDS 1
# endif
#endif
#ifndef HAVE_ASYNC_LDS
# define HAVE_ASYNC_LDS 0
#endif

#if HAVE_ASYNC_LDS
__device__ __forceinline__ void async_cp16(const unsigned short* g, unsigned short* l) {
  __builtin_amdgcn_global_load_async_to_lds_b128(
      (__attribute__((address_space(1))) v4i32*)g,
      (__attribute__((address_space(3))) v4i32*)l, 0, 0);
}
__device__ __forceinline__ void wait_async0() {
#if __has_builtin(__builtin_amdgcn_s_wait_asynccnt)
  __builtin_amdgcn_s_wait_asynccnt(0);
#else
  asm volatile("s_wait_asynccnt 0" ::: "memory");
#endif
}
#endif

// ---------------------------------------------------------------- helpers ---
__device__ __forceinline__ unsigned short f2bf(float f) {
  unsigned int u = __float_as_uint(f);
  u = (u + 0x7FFFu + ((u >> 16) & 1u)) >> 16;   // round-to-nearest-even
  return (unsigned short)u;
}

__device__ __forceinline__ float rel_c(int i) {
  if (i < 17) return (float)(i - 8);
  if (i < 22) return (float)(i - 17 - 2) * (17.0f / 5.0f);
  return 0.0f;
}

// Resolve a logical q/k/v row to (up to two) source rows + scale.
__device__ __forceinline__ void resolve_row(
    int mode, int qkv, int r, int b, int h, int d, int H,
    const float* x0, const float* x1, const float* x2,
    const float*& pA, const float*& pB, float& sc)
{
  const size_t base = (size_t)(((qkv * B_DIM + b) * H + h) * HD + d);
  pB = nullptr; sc = 1.0f;
  if (mode == 0 || qkv == 0) {
    if (r < 17)      pA = x0 + (base * 17 + r) * S_DIM;
    else if (r < 22) pA = x1 + (base * 5 + (r - 17)) * S_DIM;
    else             pA = x2 + base * S_DIM;
  } else {
    if (r == 0)      pA = x0 + (base * 17) * S_DIM;
    else if (r <= 8) { pA = x0 + (base * 17 + 2*r - 1) * S_DIM;
                       pB = x0 + (base * 17 + 2*r    ) * S_DIM; sc = 0.5f; }
    else if (r == 9) pA = x1 + (base * 5) * S_DIM;
    else if (r <= 11){ int j = r - 9;
                       pA = x1 + (base * 5 + 2*j - 1) * S_DIM;
                       pB = x1 + (base * 5 + 2*j    ) * S_DIM; sc = 0.5f; }
    else             pA = x2 + base * S_DIM;
  }
}

__device__ __forceinline__ float4 load_c4(const float4* a, const float4* b, float sc, int i) {
  float4 v = a[i];
  if (b) { float4 w = b[i]; v.x += w.x; v.y += w.y; v.z += w.z; v.w += w.w; }
  v.x *= sc; v.y *= sc; v.z *= sc; v.w *= sc;
  return v;
}

// ------------------------------------------------------- position-bias PE ---
__global__ void pe_kernel(const float* __restrict__ h_table, const float* __restrict__ h_abs,
                          const float* __restrict__ l_table, const float* __restrict__ l_abs,
                          float* __restrict__ peH, float* __restrict__ peL)
{
  __shared__ float peLf[3 * NT * NT];
  const int tid = threadIdx.x;
  for (int p = tid; p < NT * NT; p += 256) {
    int t = p / NT, u = p % NT;
    float idx = rel_c(t) - rel_c(u) + 16.0f;
    float flf = fminf(fmaxf(floorf(idx), 0.0f), 1088.0f);
    int   fl  = (int)flf;
    int   ce  = (int)fminf(fmaxf(ceilf(idx), 0.0f), 1088.0f);
    float w   = idx - flf;
    int chunk = (t < 17) ? 0 : ((t < 22) ? 1 : 2);
    peH[p] = (1.0f - w) * h_table[fl] + w * h_table[ce] + h_abs[chunk];
    for (int h = 0; h < 3; ++h)
      peLf[(h * NT + t) * NT + u] =
          (1.0f - w) * l_table[fl * 3 + h] + w * l_table[ce * 3 + h] + l_abs[chunk * 3 + h];
  }
  __syncthreads();
  for (int p = tid; p < 3 * NT * 13; p += 256) {
    int u = p % 13, t = (p / 13) % NT, h = p / (13 * NT);
    const float* row = &peLf[(h * NT + t) * NT];
    float v;
    if (u == 0)       v = row[0];
    else if (u <= 8)  v = 0.5f * (row[2*u - 1] + row[2*u]);
    else if (u == 9)  v = row[17];
    else if (u <= 11) { int j = u - 9; v = 0.5f * (row[17 + 2*j - 1] + row[17 + 2*j]); }
    else              v = row[22];
    peL[p] = v;
  }
}

// ------------------------------------------------------------- conversion ---
__global__ void f32_to_bf16_kernel(const float* __restrict__ src,
                                   unsigned short* __restrict__ dst, int n)
{
  int i = blockIdx.x * blockDim.x + threadIdx.x;
  if (i < n) dst[i] = f2bf(src[i]);
}

// --------------------------------------------------------------- attention --
__global__ void __launch_bounds__(256)
attn_kernel(const float* __restrict__ x0, const float* __restrict__ x1,
            const float* __restrict__ x2, const float* __restrict__ pe,
            unsigned short* __restrict__ att, int H, int U, int mode)
{
  const int inst = blockIdx.x;
  const int d = inst % HD;
  const int h = (inst / HD) % H;
  const int b = inst / (HD * H);
  const int tid  = threadIdx.x;
  const int wave = tid >> 5, lane = tid & 31;

  __shared__ float s_dot[NT * NT];
  __shared__ float s_rnq[NT];
  __shared__ float s_rnk[NT];

  // ---- dot products (float4 / b128 loads) ---------------------------------
  const int P = NT * U + NT + U;
  for (int p = wave; p < P; p += 8) {
    const float *qa, *qb, *ka, *kb; float qs, ks;
    if (p < NT * U) {
      int t = p / U, u = p % U;
      resolve_row(mode, 0, t, b, h, d, H, x0, x1, x2, qa, qb, qs);
      resolve_row(mode, 1, u, b, h, d, H, x0, x1, x2, ka, kb, ks);
    } else if (p < NT * U + NT) {
      int t = p - NT * U;
      resolve_row(mode, 0, t, b, h, d, H, x0, x1, x2, qa, qb, qs);
      ka = qa; kb = qb; ks = qs;
    } else {
      int u = p - NT * U - NT;
      resolve_row(mode, 1, u, b, h, d, H, x0, x1, x2, qa, qb, qs);
      ka = qa; kb = qb; ks = qs;
    }
    const float4* qa4 = (const float4*)qa; const float4* qb4 = (const float4*)qb;
    const float4* ka4 = (const float4*)ka; const float4* kb4 = (const float4*)kb;
    float acc = 0.0f;
    for (int s4 = lane; s4 < S_DIM / 4; s4 += 32) {
      float4 xv = load_c4(qa4, qb4, qs, s4);
      float4 yv = load_c4(ka4, kb4, ks, s4);
      acc = fmaf(xv.x, yv.x, acc);
      acc = fmaf(xv.y, yv.y, acc);
      acc = fmaf(xv.z, yv.z, acc);
      acc = fmaf(xv.w, yv.w, acc);
    }
    #pragma unroll
    for (int off = 16; off; off >>= 1) acc += __shfl_xor(acc, off, 32);
    if (lane == 0) {
      if (p < NT * U)           s_dot[p] = acc;
      else if (p < NT * U + NT) s_rnq[p - NT * U] = acc;
      else                      s_rnk[p - NT * U - NT] = acc;
    }
  }
  __syncthreads();
  if (tid < NT) s_rnq[tid] = 1.0f / fmaxf(sqrtf(s_rnq[tid]), 1e-12f);
  if (tid < U)  s_rnk[tid] = 1.0f / fmaxf(sqrtf(s_rnk[tid]), 1e-12f);
  __syncthreads();

  // ---- bias + softmax ------------------------------------------------------
  const float* peh = pe + (mode ? h * NT * U : 0);
  if (tid < NT) {
    float rq = s_rnq[tid], mx = -3.4e38f;
    for (int u = 0; u < U; ++u) {
      float v = s_dot[tid * U + u] * rq * s_rnk[u] + peh[tid * U + u];
      s_dot[tid * U + u] = v;
      mx = fmaxf(mx, v);
    }
    float sum = 0.0f;
    for (int u = 0; u < U; ++u) {
      float e = __expf(s_dot[tid * U + u] - mx);
      s_dot[tid * U + u] = e;
      sum += e;
    }
    float inv = 1.0f / sum;
    for (int u = 0; u < U; ++u) s_dot[tid * U + u] *= inv;
  }
  __syncthreads();

  // ---- out = attn @ v (float2 loads, packed b32 bf16 stores) ---------------
  unsigned int* att32 = (unsigned int*)att;
  for (int cpass = 0; cpass < 2; ++cpass) {
    const int s2 = tid + cpass * 256;         // float2 index within row
    float ax[NT], ay[NT];
    #pragma unroll
    for (int t = 0; t < NT; ++t) { ax[t] = 0.0f; ay[t] = 0.0f; }
    for (int u = 0; u < U; ++u) {
      const float *pa, *pb; float sc;
      resolve_row(mode, 2, u, b, h, d, H, x0, x1, x2, pa, pb, sc);
      float2 vv = ((const float2*)pa)[s2];
      if (pb) { float2 w2 = ((const float2*)pb)[s2]; vv.x += w2.x; vv.y += w2.y; }
      vv.x *= sc; vv.y *= sc;
      #pragma unroll
      for (int t = 0; t < NT; ++t) {
        float w = s_dot[t * U + u];
        ax[t] = fmaf(w, vv.x, ax[t]);
        ay[t] = fmaf(w, vv.y, ay[t]);
      }
    }
    const size_t base = (size_t)inst * NT * (S_DIM / 2) + s2;
    #pragma unroll
    for (int t = 0; t < NT; ++t) {
      unsigned int pk = (unsigned int)f2bf(ax[t]) | ((unsigned int)f2bf(ay[t]) << 16);
      att32[base + (size_t)t * (S_DIM / 2)] = pk;
    }
  }
}

// ---------------------------------------------------------- WMMA proj GEMM --
// C[m,n] = sum_k A[m,k] * W[n,k] + bias[n].  Block tile 64x64, 8 waves (4x2),
// 2 wmma per wave per K-step; double-buffered LDS with async global->LDS DMA.
__global__ void __launch_bounds__(256)
gemm_bias_kernel(const unsigned short* __restrict__ A,
                 const unsigned short* __restrict__ Wb,
                 const float* __restrict__ bias,
                 float* __restrict__ out, int mode)
{
  __shared__ __align__(16) unsigned short As[2][64][40];   // 32 used + pad
  __shared__ __align__(16) unsigned short Bs[2][64][40];

  const int tid   = threadIdx.x;
  const int lane  = tid & 31, wave = tid >> 5;
  const int waveM = wave & 3,  waveN = wave >> 2;          // 4 x 2
  const int half  = lane >> 4, l16  = lane & 15;
  const int nBase = blockIdx.x * 64;
  const int mBase = blockIdx.y * 64;

  // staging assignment: threads 0-127 -> A (64x16, k 0-15), 128-255 -> W (64x16); 16B each
  const bool loadA = tid < 128;
  const int  cc    = loadA ? tid : (tid - 128);
  const int  row   = cc >> 1, part = cc & 1;
  const unsigned short* gsrc = loadA
      ? A  + (size_t)(mBase + row) * 1024 + part * 8
      : Wb + (size_t)(nBase + row) * 1024 + part * 8;
  unsigned short* ldst0 = loadA ? &As[0][row][part * 8] : &Bs[0][row][part * 8];
  unsigned short* ldst1 = loadA ? &As[1][row][part * 8] : &Bs[1][row][part * 8];

  v8f c0 = {}, c1 = {};

#if HAVE_ASYNC_LDS
  async_cp16(gsrc, ldst0);                         // prologue: tile 0
#else
  uint4 tmp = *(const uint4*)gsrc;
  *(uint4*)ldst0 = tmp;
#endif

  for (int i = 0; i < 32; ++i) {
    const int k0n = (i + 1) * 32;
#if HAVE_ASYNC_LDS
    wait_async0();                                 // my tile-i chunks landed
    __syncthreads();                               // everyone's landed; prior readers done
    if (k0n < 1024)                                // kick off tile i+1 DMA
      async_cp16(gsrc + k0n, ((i + 1) & 1) ? ldst1 : ldst0);
#else
    __syncthreads();
    if (k0n < 1024) {
      tmp = *(const uint4*)(gsrc + k0n);           // in flight during compute
      __builtin_prefetch(gsrc + k0n + 32, 0, 1);
    }
#endif

    const int buf = i & 1;
    union { v16bf v; unsigned int u[8]; } af, bf0, bf1;
    // A 16x32 fragment: lane<16 row l16; K = half*8+{0..7} in v0-3, 16+half*8+{0..7} in v4-7
    const unsigned int* ap = (const unsigned int*)&As[buf][waveM * 16 + l16][0];
    #pragma unroll
    for (int e = 0; e < 4; ++e) af.u[e]     = ap[half * 4 + e];
    #pragma unroll
    for (int e = 0; e < 4; ++e) af.u[4 + e] = ap[8 + half * 4 + e];
    // B 32x16 fragments: lane holds column l16; K = half*8+{0..7} in v0-3, 16+half*8+{0..7} in v4-7
    const unsigned int* bp0 = (const unsigned int*)&Bs[buf][waveN * 32 + l16][0];
    const unsigned int* bp1 = (const unsigned int*)&Bs[buf][waveN * 32 + 16 + l16][0];
    #pragma unroll
    for (int e = 0; e < 8; ++e) bf0.u[e] = bp0[(e >> 2) * 8 + half * 4 + (e & 3)];
    #pragma unroll
    for (int e = 0; e < 8; ++e) bf1.u[e] = bp1[(e >> 2) * 8 + half * 4 + (e & 3)];

    c0 = __builtin_amdgcn_wmma_f32_16x16x32_bf16(false, af.v, false, bf0.v,
                                                 (short)0, c0, false, false);
    c1 = __builtin_amdgcn_wmma_f32_16x16x32_bf16(false, af.v, false, bf1.v,
                                                 (short)0, c1, false, false);

#if !HAVE_ASYNC_LDS
    if (k0n < 1024) *(uint4*)(((i + 1) & 1) ? ldst1 : ldst0) = tmp;
#endif
  }

  // ---- epilogue: bias + scatter into concatenated (B,4,64,23,S) output ----
  #pragma unroll
  for (int r = 0; r < 8; ++r) {
    const int m  = mBase + waveM * 16 + half * 8 + r;
    const int n0 = nBase + waveN * 32 + l16;
    size_t rowoff;
    if (mode == 0) {              // hifi -> channel 0
      int b = m / (HD * NT), rem = m % (HD * NT);
      int d = rem / NT, t = rem % NT;
      rowoff = ((((size_t)b * 4 + 0) * HD + d) * NT + t) * S_DIM;
    } else {                      // lofi -> channels 1..3
      int b = m / (3 * HD * NT), rem = m % (3 * HD * NT);
      int h = rem / (HD * NT); rem %= HD * NT;
      int d = rem / NT, t = rem % NT;
      rowoff = ((((size_t)b * 4 + 1 + h) * HD + d) * NT + t) * S_DIM;
    }
    out[rowoff + n0]      = c0[r] + bias[n0];
    out[rowoff + n0 + 16] = c1[r] + bias[n0 + 16];
  }
}

// ------------------------------------------------------------------ launch --
extern "C" void kernel_launch(void* const* d_in, const int* in_sizes, int n_in,
                              void* d_out, int out_size, void* d_ws, size_t ws_size,
                              hipStream_t stream) {
  (void)in_sizes; (void)n_in; (void)out_size; (void)ws_size;
  const float* xh0      = (const float*)d_in[0];
  const float* xh1      = (const float*)d_in[1];
  const float* xh2      = (const float*)d_in[2];
  const float* xl0      = (const float*)d_in[3];
  const float* xl1      = (const float*)d_in[4];
  const float* xl2      = (const float*)d_in[5];
  const float* h_table  = (const float*)d_in[6];
  const float* h_abs    = (const float*)d_in[7];
  const float* l_table  = (const float*)d_in[8];
  const float* l_abs    = (const float*)d_in[9];
  const float* h_proj_w = (const float*)d_in[10];
  const float* h_proj_b = (const float*)d_in[11];
  const float* l_proj_w = (const float*)d_in[12];
  const float* l_proj_b = (const float*)d_in[13];
  float* out = (float*)d_out;

  char* ws = (char*)d_ws;
  float*          peH  = (float*)ws;                        // 529 f32
  float*          peL  = (float*)(ws + 4096);               // 897 f32
  unsigned short* Wh   = (unsigned short*)(ws + 8192);      // 1M bf16
  unsigned short* Wl   = Wh + (size_t)1024 * 1024;          // 1M bf16
  unsigned short* attH = Wl + (size_t)1024 * 1024;          // 5888x1024 bf16
  unsigned short* attL = attH + (size_t)5888 * 1024;        // 17664x1024 bf16

  pe_kernel<<<1, 256, 0, stream>>>(h_table, h_abs, l_table, l_abs, peH, peL);
  f32_to_bf16_kernel<<<4096, 256, 0, stream>>>(h_proj_w, Wh, 1024 * 1024);
  f32_to_bf16_kernel<<<4096, 256, 0, stream>>>(l_proj_w, Wl, 1024 * 1024);

  attn_kernel<<<B_DIM * 1 * HD, 256, 0, stream>>>(xh0, xh1, xh2, peH, attH, 1, 23, 0);
  attn_kernel<<<B_DIM * 3 * HD, 256, 0, stream>>>(xl0, xl1, xl2, peL, attL, 3, 13, 1);

  gemm_bias_kernel<<<dim3(16,  92), 256, 0, stream>>>(attH, Wh, h_proj_b, out, 0);
  gemm_bias_kernel<<<dim3(16, 276), 256, 0, stream>>>(attL, Wl, l_proj_b, out, 1);
}
